// GraphConvGRUCell_10763188044360
// MI455X (gfx1250) — matrix operand order repaired and checked by reading, not compiled, not run on hardware
//
#include <hip/hip_runtime.h>
#include <hip/hip_bf16.h>
#include <math.h>

// Problem constants (from reference setup_inputs)
#define BB 8
#define NN 20000
#define EE 320000
#define II 32
#define HH 64
#define MTOT (NN * BB)            // 160000 rows of the feat matrix
#define TOT  ((size_t)BB * NN * HH)  // 10,240,000 output elements

typedef __attribute__((ext_vector_type(16))) __bf16 bf16x16;
typedef __attribute__((ext_vector_type(8)))  float  floatx8;

// ---------------------------------------------------------------------------
// K0: zero agg [N*B*H] + out_deg[N] + in_deg[N] (contiguous u32 words)
// ---------------------------------------------------------------------------
__global__ __launch_bounds__(256) void zero_kernel(unsigned* __restrict__ p, size_t nwords) {
    size_t i = (size_t)blockIdx.x * 256 + threadIdx.x;
    if (i < nwords) p[i] = 0u;
}

// ---------------------------------------------------------------------------
// K1: degree counts via u32 atomics
// ---------------------------------------------------------------------------
__global__ __launch_bounds__(256) void deg_kernel(const int* __restrict__ src,
                                                  const int* __restrict__ dst,
                                                  unsigned* __restrict__ out_deg,
                                                  unsigned* __restrict__ in_deg) {
    int e = blockIdx.x * 256 + threadIdx.x;
    if (e < EE) {
        atomicAdd(&out_deg[src[e]], 1u);
        atomicAdd(&in_deg[dst[e]], 1u);
    }
}

// ---------------------------------------------------------------------------
// K2: norms = rsqrt(max(deg,1))
// ---------------------------------------------------------------------------
__global__ __launch_bounds__(256) void norm_kernel(const unsigned* __restrict__ out_deg,
                                                   const unsigned* __restrict__ in_deg,
                                                   float* __restrict__ out_norm,
                                                   float* __restrict__ in_norm) {
    int n = blockIdx.x * 256 + threadIdx.x;
    if (n < NN) {
        unsigned od = out_deg[n]; if (od < 1u) od = 1u;
        unsigned id = in_deg[n];  if (id < 1u) id = 1u;
        out_norm[n] = rsqrtf((float)od);
        in_norm[n]  = rsqrtf((float)id);
    }
}

// ---------------------------------------------------------------------------
// K3: xr/xz/xh = x @ w + b   -> xrzh[3][B][H] (1536 floats, tiny)
// ---------------------------------------------------------------------------
__global__ __launch_bounds__(256) void xw_kernel(const float* __restrict__ x,
                                                 const float* __restrict__ wr, const float* __restrict__ br,
                                                 const float* __restrict__ wz, const float* __restrict__ bz,
                                                 const float* __restrict__ wh, const float* __restrict__ bh,
                                                 float* __restrict__ xrzh) {
    int idx = blockIdx.x * 256 + threadIdx.x;       // 0 .. 3*B*H-1
    if (idx >= 3 * BB * HH) return;
    int k = idx & (HH - 1);
    int b = (idx >> 6) & (BB - 1);
    int g = idx >> 9;
    const float* w    = (g == 0) ? wr : ((g == 1) ? wz : wh);
    const float* bias = (g == 0) ? br : ((g == 1) ? bz : bh);
    float s = bias[k];
    #pragma unroll
    for (int i = 0; i < II; ++i) s += x[b * II + i] * w[i * HH + k];
    xrzh[idx] = s;
}

// ---------------------------------------------------------------------------
// K4: pre-pack gcn_w (64x64 f32) into bf16 WMMA B-fragments.
// Layout: bfrag[ct][kc][lane][j], ct=col-tile(4), kc=k-chunk(2), lane(32), j(16).
// ISA 16-bit B 32x16: lane<16 -> N=lane, K = kc*32 + (j<8 ? j : 16+j-8);
//                     lane>=16 -> N=lane-16, K = kc*32 + 8 + (j<8 ? j : 16+j-8).
// ---------------------------------------------------------------------------
__global__ __launch_bounds__(256) void bfrag_kernel(const float* __restrict__ gcn_w,
                                                    __bf16* __restrict__ bfrag) {
    int idx = blockIdx.x * 256 + threadIdx.x;       // 0 .. 4095
    if (idx >= 4 * 2 * 32 * 16) return;
    int j    = idx & 15;
    int lane = (idx >> 4) & 31;
    int kc   = (idx >> 9) & 1;
    int ct   = idx >> 10;
    int col  = ct * 16 + (lane & 15);
    int K    = kc * 32 + (lane >> 4) * 8 + (j & 7) + (j >> 3) * 16;
    bfrag[idx] = (__bf16)gcn_w[K * HH + col];
}

// ---------------------------------------------------------------------------
// K5: feat[m,k] = out_norm[m>>3] * (h_prev_row(m) @ gcn_w)  via bf16 WMMA.
// m = n*8 + b; h_prev_row(m) = h_prev[b, n, :]. One wave = 16-row x 64-col tile.
// ---------------------------------------------------------------------------
__global__ __launch_bounds__(256) void feat_wmma_kernel(const float* __restrict__ h_prev,
                                                        const float* __restrict__ out_norm,
                                                        const __bf16* __restrict__ bfrag,
                                                        float* __restrict__ feat) {
    const int lane = threadIdx.x & 31;
    const int wave = threadIdx.x >> 5;
    const int tile = blockIdx.x * 8 + wave;          // 16-row tile index (exact: 10000 tiles)
    const int rowA = lane & 15;                      // A-matrix row for this lane
    const int hi   = lane >> 4;                      // half-wave selector
    const int m    = tile * 16 + rowA;
    const int node = m >> 3;
    const int b    = m & 7;
    const float* __restrict__ arow = h_prev + ((size_t)b * NN + node) * HH;
    const int koff = hi * 8;                         // ISA A layout K offset per half-wave

    // Build A fragments for K-chunks 0 (K=0..31) and 1 (K=32..63)
    bf16x16 afr[2];
    #pragma unroll
    for (int kc = 0; kc < 2; ++kc) {
        const float* p = arow + kc * 32 + koff;
        float4 f0 = *(const float4*)(p + 0);         // K = base+0..3
        float4 f1 = *(const float4*)(p + 4);         // K = base+4..7
        float4 f2 = *(const float4*)(p + 16);        // K = base+16..19
        float4 f3 = *(const float4*)(p + 20);        // K = base+20..23
        bf16x16 a;
        a[0]  = (__bf16)f0.x; a[1]  = (__bf16)f0.y; a[2]  = (__bf16)f0.z; a[3]  = (__bf16)f0.w;
        a[4]  = (__bf16)f1.x; a[5]  = (__bf16)f1.y; a[6]  = (__bf16)f1.z; a[7]  = (__bf16)f1.w;
        a[8]  = (__bf16)f2.x; a[9]  = (__bf16)f2.y; a[10] = (__bf16)f2.z; a[11] = (__bf16)f2.w;
        a[12] = (__bf16)f3.x; a[13] = (__bf16)f3.y; a[14] = (__bf16)f3.z; a[15] = (__bf16)f3.w;
        afr[kc] = a;
    }

    #pragma unroll
    for (int ct = 0; ct < 4; ++ct) {                 // 4 col-tiles of 16
        bf16x16 b0 = *(const bf16x16*)(bfrag + ((size_t)((ct * 2 + 0) * 32 + lane)) * 16);
        bf16x16 b1 = *(const bf16x16*)(bfrag + ((size_t)((ct * 2 + 1) * 32 + lane)) * 16);
        floatx8 c = {};
        c = __builtin_amdgcn_wmma_f32_16x16x32_bf16(false, afr[0], false, b0, (short)0, c, false, false);
        c = __builtin_amdgcn_wmma_f32_16x16x32_bf16(false, afr[1], false, b1, (short)0, c, false, false);
        // C layout: VGPR r -> row = hi*8 + r, col = lane&15
        #pragma unroll
        for (int r = 0; r < 8; ++r) {
            int mg = tile * 16 + hi * 8 + r;
            float on = out_norm[mg >> 3];
            feat[(size_t)mg * HH + ct * 16 + (lane & 15)] = c[r] * on;
        }
    }
}

// ---------------------------------------------------------------------------
// K6: scatter-add: agg[dst[e], :, :] += feat[src[e], :, :]  (512 floats/edge)
// feat & agg both fit in the 192MB L2 -> atomics stay on-chip.
// ---------------------------------------------------------------------------
__global__ __launch_bounds__(512) void scatter_kernel(const int* __restrict__ src,
                                                      const int* __restrict__ dst,
                                                      const float* __restrict__ feat,
                                                      float* __restrict__ agg) {
    int e = blockIdx.x;
    int s = src[e];
    int d = dst[e];
    const float* fp = feat + (size_t)s * (BB * HH);
    float* ap = agg + (size_t)d * (BB * HH);
    int t = threadIdx.x;                             // 0..511 = b*64+k
    atomicAdd(&ap[t], fp[t]);
}

// ---------------------------------------------------------------------------
// K7: fused GRU update: a = in_norm*agg + gcn_b; r,z = sigmoid; h~ = tanh;
//     h_t = (1-z)*h_prev + z*h~   (output layout [B,N,H])
// ---------------------------------------------------------------------------
__global__ __launch_bounds__(256) void final_kernel(const float* __restrict__ h_prev,
                                                    const float* __restrict__ agg,
                                                    const float* __restrict__ in_norm,
                                                    const float* __restrict__ xrzh,
                                                    const float* __restrict__ gcn_b,
                                                    float* __restrict__ out) {
    size_t idx = (size_t)blockIdx.x * 256 + threadIdx.x;
    if (idx >= TOT) return;
    int k = (int)(idx & (HH - 1));
    size_t nb = idx >> 6;
    int n = (int)(nb % NN);
    int b = (int)(nb / NN);
    float a  = in_norm[n] * agg[((size_t)n * BB + b) * HH + k] + gcn_b[k];
    float xr = xrzh[          b * HH + k];
    float xz = xrzh[ 512 +    b * HH + k];
    float xh = xrzh[1024 +    b * HH + k];
    float r  = 1.0f / (1.0f + expf(-(xr + a)));
    float z  = 1.0f / (1.0f + expf(-(xz + a)));
    float ht = tanhf(xh + r * a);
    float hp = h_prev[idx];
    out[idx] = (1.0f - z) * hp + z * ht;
}

// ---------------------------------------------------------------------------
extern "C" void kernel_launch(void* const* d_in, const int* in_sizes, int n_in,
                              void* d_out, int out_size, void* d_ws, size_t ws_size,
                              hipStream_t stream) {
    const float* x      = (const float*)d_in[0];
    const float* h_prev = (const float*)d_in[1];
    const int*   src    = (const int*)d_in[2];
    const int*   dst    = (const int*)d_in[3];
    const float* w_r_w  = (const float*)d_in[4];
    const float* w_r_b  = (const float*)d_in[5];
    const float* w_z_w  = (const float*)d_in[6];
    const float* w_z_b  = (const float*)d_in[7];
    const float* w_h_w  = (const float*)d_in[8];
    const float* w_h_b  = (const float*)d_in[9];
    const float* gcn_w  = (const float*)d_in[10];
    const float* gcn_b  = (const float*)d_in[11];

    // Workspace layout (words): agg | out_deg | in_deg | out_norm | in_norm | xrzh | bfrag
    const size_t AGG_W = TOT;                        // 10,240,000 floats
    float*    ws       = (float*)d_ws;
    float*    agg      = ws;
    unsigned* out_deg  = (unsigned*)(ws + AGG_W);
    unsigned* in_deg   = out_deg + NN;
    float*    out_norm = (float*)(in_deg + NN);
    float*    in_norm  = out_norm + NN;
    float*    xrzh     = in_norm + NN;
    __bf16*   bfrag    = (__bf16*)(xrzh + 3 * BB * HH);   // 32B-aligned offset
    float*    feat     = (float*)d_out;              // reuse d_out as feat scratch

    // 1) zero agg + degree counters (contiguous)
    size_t zwords = AGG_W + 2 * (size_t)NN;
    zero_kernel<<<dim3((unsigned)((zwords + 255) / 256)), dim3(256), 0, stream>>>((unsigned*)ws, zwords);

    // 2) degrees
    deg_kernel<<<dim3((EE + 255) / 256), dim3(256), 0, stream>>>(src, dst, out_deg, in_deg);

    // 3) norms
    norm_kernel<<<dim3((NN + 255) / 256), dim3(256), 0, stream>>>(out_deg, in_deg, out_norm, in_norm);

    // 4) x projections
    xw_kernel<<<dim3((3 * BB * HH + 255) / 256), dim3(256), 0, stream>>>(
        x, w_r_w, w_r_b, w_z_w, w_z_b, w_h_w, w_h_b, xrzh);

    // 5) pack gcn_w into bf16 WMMA B-fragments
    bfrag_kernel<<<dim3(16), dim3(256), 0, stream>>>(gcn_w, bfrag);

    // 6) feat = out_norm * (h_prev @ gcn_w)   [WMMA, 10000 tiles / 8 waves per block]
    feat_wmma_kernel<<<dim3(MTOT / 16 / 8), dim3(256), 0, stream>>>(h_prev, out_norm, bfrag, feat);

    // 7) edge scatter-add into agg
    scatter_kernel<<<dim3(EE), dim3(BB * HH), 0, stream>>>(src, dst, feat, agg);

    // 8) fused GRU elementwise -> d_out
    final_kernel<<<dim3((unsigned)((TOT + 255) / 256)), dim3(256), 0, stream>>>(
        h_prev, agg, in_norm, xrzh, gcn_b, (float*)d_out);
}
